// SlidingWindowGQACausalSelfAttention_80556406603856
// MI455X (gfx1250) — compile-verified, matrix operand order checked by reading, not run on hardware
//
#include <hip/hip_runtime.h>
#include <hip/hip_bf16.h>

#define B_    2
#define S_    2048
#define DIM_  2048
#define H_    16
#define KVH_  4
#define HD_   128
#define WIN_  512
#define EPS_  1.1920928955078125e-07f

typedef __bf16 v8bf  __attribute__((ext_vector_type(8)));
typedef __bf16 v16bf __attribute__((ext_vector_type(16)));
typedef float  v8f   __attribute__((ext_vector_type(8)));

__device__ __forceinline__ v8bf ld8(const __hip_bfloat16* p) {
    return *reinterpret_cast<const v8bf*>(p);
}
__device__ __forceinline__ v16bf cat8(v8bf lo, v8bf hi) {
    return __builtin_shufflevector(lo, hi, 0,1,2,3,4,5,6,7,8,9,10,11,12,13,14,15);
}

// async copy of 16 bytes global -> LDS (ASYNCcnt-tracked)
__device__ __forceinline__ void async_cp16(const __hip_bfloat16* src, __hip_bfloat16* lds_dst) {
    unsigned dst = (unsigned)(uintptr_t)lds_dst;             // low 32 bits = LDS offset
    unsigned long long g = (unsigned long long)(uintptr_t)src;
    asm volatile("global_load_async_to_lds_b128 %0, %1, off"
                 :: "v"(dst), "v"(g) : "memory");
}

// ---------------------------------------------------------------- casts
__global__ void cast_f32_bf16(const float* __restrict__ in,
                              __hip_bfloat16* __restrict__ out, int n) {
    int i = blockIdx.x * 256 + threadIdx.x;
    if (i < n) out[i] = __float2bfloat16(in[i]);
}

// Vf: [B*S, KVH*HD] f32  ->  Vt: [B, KVH, HD, S] bf16 (transposed for PV B-frags)
__global__ void vcast_transpose(const float* __restrict__ Vf,
                                __hip_bfloat16* __restrict__ Vt) {
    int i = blockIdx.x * 256 + threadIdx.x;           // over B*S*KVH*HD
    if (i >= B_ * S_ * KVH_ * HD_) return;
    int col = i & (KVH_ * HD_ - 1);                   // 512
    int row = i >> 9;
    int s = row & (S_ - 1);
    int b = row >> 11;
    int kv = col >> 7;
    int hd = col & (HD_ - 1);
    Vt[(((size_t)b * KVH_ + kv) * HD_ + hd) * S_ + s] = __float2bfloat16(Vf[i]);
}

// ---------------------------------------------------------------- generic WMMA GEMM
// C[M,N] = A[M,K] * B[N,K]^T   (A,B bf16 row-major over K; C f32)
__global__ __launch_bounds__(128)
void gemm_bf16_wmma(const __hip_bfloat16* __restrict__ A,
                    const __hip_bfloat16* __restrict__ Bm,
                    float* __restrict__ C, int M, int N, int K) {
    const int lane = threadIdx.x & 31;
    const int wave = threadIdx.x >> 5;
    const int sub  = lane & 15;
    const int hi   = lane >> 4;
    const int m0 = blockIdx.y * 64 + wave * 16;
    const int n0 = blockIdx.x * 64;
    if (m0 >= M) return;

    v8f acc[4] = {};
    const __hip_bfloat16* arow = A + (size_t)(m0 + sub) * K;

    for (int kb = 0; kb < K; kb += 32) {
        __builtin_prefetch(arow + kb + 512, 0, 1);     // global_prefetch_b8
        v16bf af = cat8(ld8(arow + kb + hi * 8), ld8(arow + kb + 16 + hi * 8));
#pragma unroll
        for (int t = 0; t < 4; ++t) {
            const __hip_bfloat16* brow = Bm + (size_t)(n0 + t * 16 + sub) * K;
            v16bf bf = cat8(ld8(brow + kb + hi * 8), ld8(brow + kb + 16 + hi * 8));
            acc[t] = __builtin_amdgcn_wmma_f32_16x16x32_bf16(
                false, af, false, bf, (short)0, acc[t], false, false);
        }
    }
#pragma unroll
    for (int t = 0; t < 4; ++t)
#pragma unroll
        for (int r = 0; r < 8; ++r) {
            int row = m0 + r + hi * 8;
            int col = n0 + t * 16 + sub;
            C[(size_t)row * N + col] = acc[t][r];
        }
}

// ---------------------------------------------------------------- RMSNorm + RoPE (+gain)
// in: [B*S, nheads*HD] f32 ; out: [B, nheads, S, HD] bf16
__global__ __launch_bounds__(128)
void rmsnorm_rope(const float* __restrict__ in, __hip_bfloat16* __restrict__ out,
                  const float* __restrict__ gain, int nheads) {
    const int h   = blockIdx.y;
    const int row = blockIdx.x;                 // b*S + s
    const int s   = row & (S_ - 1);
    const int b   = row >> 11;
    const int tid = threadIdx.x;                // 0..127

    float v = in[(size_t)row * (nheads * HD_) + h * HD_ + tid];
    float p = v * v;
#pragma unroll
    for (int m = 16; m >= 1; m >>= 1) p += __shfl_xor(p, m, 32);

    __shared__ float red[4];
    __shared__ float sval[128];
    if ((tid & 31) == 0) red[tid >> 5] = p;
    sval[tid] = v;
    __syncthreads();
    float inv = rsqrtf((red[0] + red[1] + red[2] + red[3]) * (1.0f / HD_) + EPS_);

    float val     = v * inv;
    float partner = sval[tid ^ 64] * inv;
    int   idx     = tid & 63;
    // inv_freq = base^(-idx/64) = exp(-idx * ln(10000)/64)
    float f = (float)s * __expf((float)idx * (-9.210340371976184f / 64.0f));
    float sn, cs;
    __sincosf(f, &sn, &cs);
    float o = (tid < 64) ? (val * cs + partner * sn)      // x1*c + x2*s
                         : (-partner * sn + val * cs);    // -x1*s + x2*c
    if (gain) o *= gain[h];
    out[(((size_t)b * nheads + h) * S_ + s) * HD_ + tid] = __float2bfloat16(o);
}

// ---------------------------------------------------------------- flash attention (window 512)
// Qr: [B,H,S,HD], Kr: [B,KVH,S,HD], Vt: [B,KVH,HD,S] (bf16) -> Y: [B*S, DIM] bf16
// Block = 128 threads (4 waves); K/V tiles staged block-cooperatively into
// double-buffered LDS via global_load_async_to_lds_b128, pipelined on ASYNCcnt.
__global__ __launch_bounds__(128)
void attn_kernel(const __hip_bfloat16* __restrict__ Qr,
                 const __hip_bfloat16* __restrict__ Kr,
                 const __hip_bfloat16* __restrict__ Vt,
                 __hip_bfloat16* __restrict__ Y) {
    const int tid  = threadIdx.x;
    const int lane = tid & 31;
    const int wave = tid >> 5;
    const int sub  = lane & 15;
    const int hi   = lane >> 4;
    const int bh   = blockIdx.x;                 // b*H + h
    const int b    = bh / H_;
    const int h    = bh % H_;
    const int kvh  = h / (H_ / KVH_);
    const int qb0  = blockIdx.y * 64;            // block-uniform query base
    const int qt0  = qb0 + wave * 16;            // this wave's 16-row tile

    __shared__ __align__(16) __hip_bfloat16 Ktile[2][32][128];  // [buf][key][hd]
    __shared__ __align__(16) __hip_bfloat16 Vtile[2][128][32];  // [buf][hd][key]
    __shared__ __align__(16) __hip_bfloat16 pbuf[4][16][40];

    const __hip_bfloat16* kbase = Kr + (size_t)(b * KVH_ + kvh) * S_ * HD_;
    const __hip_bfloat16* vbase = Vt + (size_t)(b * KVH_ + kvh) * HD_ * S_;

    // Q tile as 4 A-fragments (K chunks of 32 over HD=128)
    const __hip_bfloat16* qbase = Qr + ((size_t)bh * S_ + qt0 + sub) * HD_;
    v16bf aq[4];
#pragma unroll
    for (int c = 0; c < 4; ++c)
        aq[c] = cat8(ld8(qbase + 32 * c + hi * 8), ld8(qbase + 32 * c + 16 + hi * 8));

    v8f acc[8] = {};
    float mrow[8], lrow[8];
#pragma unroll
    for (int r = 0; r < 8; ++r) { mrow[r] = -INFINITY; lrow[r] = 0.0f; }

    // issue one 32-key K+V tile (8 async b128 per thread, 1024 total per block)
    auto issue_tile = [&](int kb, int buf) {
#pragma unroll
        for (int i = 0; i < 4; ++i) {
            int c = tid + 128 * i;                        // 0..511
            int krow = c >> 4, kcol = (c & 15) * 8;       // 32 rows x 16 chunks
            async_cp16(kbase + (size_t)(kb + krow) * HD_ + kcol,
                       &Ktile[buf][krow][kcol]);
            int vrow = c >> 2, vcol = (c & 3) * 8;        // 128 rows x 4 chunks
            async_cp16(vbase + (size_t)vrow * S_ + kb + vcol,
                       &Vtile[buf][vrow][vcol]);
        }
    };

    int kb_lo = qb0 - (WIN_ - 1);
    if (kb_lo < 0) kb_lo = 0;
    kb_lo &= ~31;
    const int kend  = qb0 + 63;                   // block-union upper bound
    const float scale = 0.088388347648318447f;    // 1/sqrt(128)

    issue_tile(kb_lo, 0);

    int ibuf = 0;
    for (int kb = kb_lo; kb <= kend; kb += 32, ibuf ^= 1) {
        const bool hasNext = (kb + 32 <= kend);   // block-uniform
        if (hasNext) {
            issue_tile(kb + 32, ibuf ^ 1);
            asm volatile("s_wait_asynccnt 0x8" ::: "memory");  // current tile done
        } else {
            asm volatile("s_wait_asynccnt 0x0" ::: "memory");
        }
        __syncthreads();                          // tile visible to all waves

        // ---- scores: two 16-key tiles, 4 WMMAs each over HD (B-frags from LDS)
        v8f sc[2];
#pragma unroll
        for (int j = 0; j < 2; ++j) {
            v8f s0 = {};
            const __hip_bfloat16* krow = &Ktile[ibuf][16 * j + sub][0];
#pragma unroll
            for (int c = 0; c < 4; ++c) {
                v16bf bk = cat8(ld8(krow + 32 * c + hi * 8),
                                ld8(krow + 32 * c + 16 + hi * 8));
                s0 = __builtin_amdgcn_wmma_f32_16x16x32_bf16(
                    false, aq[c], false, bk, (short)0, s0, false, false);
            }
            sc[j] = s0;
        }
        // ---- mask + online softmax (rows live in VGPR index r, 16-lane halves)
#pragma unroll
        for (int r = 0; r < 8; ++r) {
            int row = qt0 + r + 8 * hi;
            float m_old = mrow[r];
            float s0 = sc[0][r] * scale, s1 = sc[1][r] * scale;
            int c0 = kb + sub, c1 = kb + 16 + sub;
            bool v0 = (c0 <= row) && (c0 > row - WIN_);
            bool v1 = (c1 <= row) && (c1 > row - WIN_);
            s0 = v0 ? s0 : -INFINITY;
            s1 = v1 ? s1 : -INFINITY;
            float mx = fmaxf(s0, s1);
#pragma unroll
            for (int msk = 8; msk >= 1; msk >>= 1) mx = fmaxf(mx, __shfl_xor(mx, msk, 32));
            float m_new = fmaxf(m_old, mx);
            float corr = (m_new == -INFINITY) ? 1.0f : __expf(m_old - m_new);
            float p0 = (m_new == -INFINITY) ? 0.0f : __expf(s0 - m_new);
            float p1 = (m_new == -INFINITY) ? 0.0f : __expf(s1 - m_new);
            float ps = p0 + p1;
#pragma unroll
            for (int msk = 8; msk >= 1; msk >>= 1) ps += __shfl_xor(ps, msk, 32);
            lrow[r] = lrow[r] * corr + ps;
            mrow[r] = m_new;
#pragma unroll
            for (int t = 0; t < 8; ++t) acc[t][r] *= corr;
            // C-layout -> LDS (for A-layout reload)
            pbuf[wave][r + 8 * hi][sub]      = __float2bfloat16(p0);
            pbuf[wave][r + 8 * hi][16 + sub] = __float2bfloat16(p1);
        }
        asm volatile("s_wait_dscnt 0x0" ::: "memory");   // wave-internal LDS ordering
        __builtin_amdgcn_wave_barrier();
        v16bf pa = cat8(*reinterpret_cast<const v8bf*>(&pbuf[wave][sub][hi * 8]),
                        *reinterpret_cast<const v8bf*>(&pbuf[wave][sub][16 + hi * 8]));
        // ---- PV: out(16x128) += P(16x32) x V(32x128)  (B-frags from LDS Vtile)
#pragma unroll
        for (int t = 0; t < 8; ++t) {
            const __hip_bfloat16* vrow = &Vtile[ibuf][t * 16 + sub][0];
            v16bf bv = cat8(ld8(vrow + hi * 8), ld8(vrow + 16 + hi * 8));
            acc[t] = __builtin_amdgcn_wmma_f32_16x16x32_bf16(
                false, pa, false, bv, (short)0, acc[t], false, false);
        }
        __syncthreads();                          // readers done before buffer reuse
    }
    // ---- normalize + store bf16 into [B*S, DIM]
#pragma unroll
    for (int r = 0; r < 8; ++r) {
        float invl = (lrow[r] > 0.0f) ? (1.0f / lrow[r]) : 0.0f;
        int srow = qt0 + r + 8 * hi;
        size_t orow = ((size_t)b * S_ + srow) * DIM_;
#pragma unroll
        for (int t = 0; t < 8; ++t)
            Y[orow + h * HD_ + t * 16 + sub] = __float2bfloat16(acc[t][r] * invl);
    }
}

// ---------------------------------------------------------------- launch
extern "C" void kernel_launch(void* const* d_in, const int* in_sizes, int n_in,
                              void* d_out, int out_size, void* d_ws, size_t ws_size,
                              hipStream_t stream) {
    const float* x  = (const float*)d_in[0];
    const float* Wq = (const float*)d_in[1];
    const float* Wk = (const float*)d_in[2];
    const float* Wv = (const float*)d_in[3];
    const float* Wo = (const float*)d_in[4];
    const float* qg = (const float*)d_in[5];
    float* out = (float*)d_out;

    const size_t MS = (size_t)B_ * S_;          // 4096
    char* ws = (char*)d_ws;
    size_t off = 0;
    auto alloc = [&](size_t bytes) { void* p = ws + off; off = (off + bytes + 255) & ~(size_t)255; return p; };

    __hip_bfloat16* xb  = (__hip_bfloat16*)alloc(MS * DIM_ * 2);
    __hip_bfloat16* Wqb = (__hip_bfloat16*)alloc((size_t)DIM_ * DIM_ * 2);
    __hip_bfloat16* Wkb = (__hip_bfloat16*)alloc((size_t)KVH_ * HD_ * DIM_ * 2);
    __hip_bfloat16* Wvb = (__hip_bfloat16*)alloc((size_t)KVH_ * HD_ * DIM_ * 2);
    __hip_bfloat16* Wob = (__hip_bfloat16*)alloc((size_t)DIM_ * DIM_ * 2);
    float* Qf = (float*)alloc(MS * DIM_ * 4);
    float* Kf = (float*)alloc(MS * KVH_ * HD_ * 4);
    float* Vf = (float*)alloc(MS * KVH_ * HD_ * 4);
    __hip_bfloat16* Qr  = (__hip_bfloat16*)alloc(MS * DIM_ * 2);
    __hip_bfloat16* Kr  = (__hip_bfloat16*)alloc(MS * KVH_ * HD_ * 2);
    __hip_bfloat16* Vtb = (__hip_bfloat16*)alloc(MS * KVH_ * HD_ * 2);
    __hip_bfloat16* Y   = (__hip_bfloat16*)Qf;   // alias: Qf dead before attention

    auto blocks = [](size_t n) { return (unsigned)((n + 255) / 256); };

    // 1) casts
    cast_f32_bf16<<<blocks(MS * DIM_), 256, 0, stream>>>(x, xb, (int)(MS * DIM_));
    cast_f32_bf16<<<blocks((size_t)DIM_ * DIM_), 256, 0, stream>>>(Wq, Wqb, DIM_ * DIM_);
    cast_f32_bf16<<<blocks((size_t)512 * DIM_), 256, 0, stream>>>(Wk, Wkb, 512 * DIM_);
    cast_f32_bf16<<<blocks((size_t)512 * DIM_), 256, 0, stream>>>(Wv, Wvb, 512 * DIM_);
    cast_f32_bf16<<<blocks((size_t)DIM_ * DIM_), 256, 0, stream>>>(Wo, Wob, DIM_ * DIM_);

    // 2) QKV projections (C = x * W^T)
    gemm_bf16_wmma<<<dim3(DIM_ / 64, MS / 64), 128, 0, stream>>>(xb, Wqb, Qf, (int)MS, DIM_, DIM_);
    gemm_bf16_wmma<<<dim3(512 / 64,  MS / 64), 128, 0, stream>>>(xb, Wkb, Kf, (int)MS, 512, DIM_);
    gemm_bf16_wmma<<<dim3(512 / 64,  MS / 64), 128, 0, stream>>>(xb, Wvb, Vf, (int)MS, 512, DIM_);

    // 3) RMSNorm + RoPE (+gain for Q); V transpose-cast
    rmsnorm_rope<<<dim3((unsigned)MS, H_),   128, 0, stream>>>(Qf, Qr, qg, H_);
    rmsnorm_rope<<<dim3((unsigned)MS, KVH_), 128, 0, stream>>>(Kf, Kr, nullptr, KVH_);
    vcast_transpose<<<blocks(MS * KVH_ * HD_), 256, 0, stream>>>(Vf, Vtb);

    // 4) sliding-window attention (async-LDS pipelined)
    attn_kernel<<<dim3(B_ * H_, S_ / 64), 128, 0, stream>>>(Qr, Kr, Vtb, Y);

    // 5) output projection -> f32 d_out
    gemm_bf16_wmma<<<dim3(DIM_ / 64, MS / 64), 128, 0, stream>>>(Y, Wob, out, (int)MS, DIM_, DIM_);
}
